// DynamicConv2d_44178033607355
// MI455X (gfx1250) — matrix operand order repaired and checked by reading, not compiled, not run on hardware
//
#include <hip/hip_runtime.h>
#include <hip/hip_bf16.h>
#include <math.h>

typedef __attribute__((ext_vector_type(16))) _Float16 v16h;
typedef __attribute__((ext_vector_type(8)))  float    v8f;
typedef __attribute__((ext_vector_type(4)))  unsigned int v4u;
typedef __attribute__((ext_vector_type(4)))  int v4i;

typedef __attribute__((address_space(1))) v4i global_v4i;  // AS1 (device/global)
typedef __attribute__((address_space(3))) v4i lds_v4i;     // AS3 (LDS)

union AFrag { v16h h; v4u u[2]; };

#define BATCH 32
#define CCH   128
#define HH    64
#define WW    64
#define NK    4
#define NTAP  9

#define HAS_ASYNC_LDS __has_builtin(__builtin_amdgcn_global_load_async_to_lds_b128)

// 16-byte global -> LDS copy: async DMA on CDNA5 (ASYNCcnt) if available,
// else register bounce (LOADcnt + DScnt, synchronized by __syncthreads).
__device__ __forceinline__ void copy_b128_to_lds(const _Float16* gp, _Float16* lp) {
#if HAS_ASYNC_LDS
    __builtin_amdgcn_global_load_async_to_lds_b128(
        (global_v4i*)(v4i*)gp,
        (lds_v4i*)(v4i*)lp, 0, 0);
#else
    *(v4u*)lp = *(const v4u*)gp;
#endif
}

__device__ __forceinline__ void wait_async_lds() {
#if HAS_ASYNC_LDS
#if __has_builtin(__builtin_amdgcn_s_wait_asynccnt)
    __builtin_amdgcn_s_wait_asynccnt(0);
#else
    asm volatile("s_wait_asynccnt 0x0" ::: "memory");
#endif
#endif
}

// ---------------------------------------------------------------------------
// Kernel 1: routing — GAP over HxW, linear(K=4), sigmoid, softmax -> coeffs[B][4]
// ---------------------------------------------------------------------------
__global__ void dyn_routing_kernel(const float* __restrict__ x,
                                   const float* __restrict__ fc_w,
                                   const float* __restrict__ fc_b,
                                   float* __restrict__ coeffs) {
    const int b   = blockIdx.x;
    const int tid = threadIdx.x;
    const int c    = tid & 127;
    const int part = tid >> 7;

    const float* xp = x + (((size_t)b * CCH + c) * (HH * WW)) + part * 2048;
    float s = 0.f;
    #pragma unroll 4
    for (int i = 0; i < 512; ++i) {
        float4 v = ((const float4*)xp)[i];
        s += v.x + v.y + v.z + v.w;
    }

    __shared__ float red[256];
    __shared__ float pooled[128];
    __shared__ float lg[4];
    red[tid] = s;
    __syncthreads();
    if (tid < 128) pooled[tid] = (red[tid] + red[tid + 128]) * (1.f / 4096.f);
    __syncthreads();
    if (tid < NK) {
        float d = fc_b[tid];
        for (int cc = 0; cc < 128; ++cc) d += pooled[cc] * fc_w[tid * 128 + cc];
        lg[tid] = 1.f / (1.f + expf(-d));
    }
    __syncthreads();
    if (tid == 0) {
        float m = lg[0];
        for (int k = 1; k < NK; ++k) m = lg[k] > m ? lg[k] : m;
        float e[NK], sum = 0.f;
        for (int k = 0; k < NK; ++k) { e[k] = expf(lg[k] - m); sum += e[k]; }
        float inv = 1.f / sum;
        for (int k = 0; k < NK; ++k) coeffs[b * NK + k] = e[k] * inv;
    }
}

// ---------------------------------------------------------------------------
// Kernel 2: blend 4 weight banks -> f16, pre-transposed Ah[b][tap][co][ci]
// ---------------------------------------------------------------------------
__global__ void dyn_mix_kernel(const float* __restrict__ kernels,
                               const float* __restrict__ coeffs,
                               _Float16* __restrict__ Ah) {
    const size_t idx = (size_t)blockIdx.x * 256 + threadIdx.x;
    const int ci   = (int)(idx & 127);
    const int co   = (int)((idx >> 7) & 127);
    const int rest = (int)(idx >> 14);
    const int tap  = rest % NTAP;
    const int b    = rest / NTAP;

    const float c0 = coeffs[b * NK + 0];
    const float c1 = coeffs[b * NK + 1];
    const float c2 = coeffs[b * NK + 2];
    const float c3 = coeffs[b * NK + 3];

    const size_t base = ((size_t)co * CCH + ci) * NTAP + tap;
    const size_t kstr = (size_t)CCH * CCH * NTAP;
    float v = c0 * kernels[base]
            + c1 * kernels[base + kstr]
            + c2 * kernels[base + 2 * kstr]
            + c3 * kernels[base + 3 * kstr];
    Ah[idx] = (_Float16)v;
}

// ---------------------------------------------------------------------------
// Kernel 3: x (f32 NCHW) -> xh (f16 channel-last [b][h][w][ci])
// ---------------------------------------------------------------------------
__global__ void dyn_xcvt_kernel(const float* __restrict__ x,
                                _Float16* __restrict__ xh) {
    const size_t idx = (size_t)blockIdx.x * 256 + threadIdx.x;
    const int w  = (int)(idx & 63);
    const int h  = (int)((idx >> 6) & 63);
    const int ci = (int)((idx >> 12) & 127);
    const int b  = (int)(idx >> 19);
    float v = x[idx];
    xh[((((size_t)b * HH + h) * WW + w) * CCH) + ci] = (_Float16)v;
}

// ---------------------------------------------------------------------------
// Kernel 4: implicit-GEMM dynamic conv, v_wmma_f32_16x16x32_f16.
// grid = B*H; 256 threads = 8 waves in a 4(M) x 2(N) arrangement.
// Wave (wm,wn): co in [32*wm, 32*wm+32) x pixels [32*wn, 32*wn+32)
//   -> 2x2 accumulators of 16x16, giving 2-way register reuse on A and B.
// K loop: 4 chunks of 32 ci (double-buffered async LDS strips) x 9 taps.
// ---------------------------------------------------------------------------
__global__ void __launch_bounds__(256, 2)
dyn_conv_wmma_kernel(const _Float16* __restrict__ xh,
                     const _Float16* __restrict__ Ah,
                     float* __restrict__ out) {
    const int b  = blockIdx.x >> 6;
    const int h  = blockIdx.x & 63;
    const int tid  = threadIdx.x;
    const int wave = tid >> 5;
    const int lane = tid & 31;
    const int wm   = wave & 3;      // 4 waves along co
    const int wn   = wave >> 2;     // 2 waves along pixels
    const int cob  = wm * 32;
    const int pb   = wn * 32;
    const int m    = lane & 15;
    const int hlf  = lane >> 4;

    // double-buffered 3-row x 64 x 32-channel f16 strips (2 x 12 KB)
    __shared__ __align__(16) _Float16 strip[2][3][WW][32];

    v8f acc[2][2];
    #pragma unroll
    for (int i = 0; i < 2; ++i)
        #pragma unroll
        for (int j = 0; j < 2; ++j)
            #pragma unroll
            for (int e = 0; e < 8; ++e) acc[i][j][e] = 0.f;

    const v4u vzero = {0u, 0u, 0u, 0u};
    const bool rok[3] = { h > 0, true, h < HH - 1 };   // per-kh row validity

    // stage one ci-chunk: 3 rows x 64 w x 32 ci = 768 b128 chunks, 3/thread
    auto stage = [&](int buf, int ci0) {
        #pragma unroll
        for (int i = 0; i < 3; ++i) {
            const int chunk = tid + i * 256;
            const int r   = chunk >> 8;
            const int rem = chunk & 255;
            const int w   = rem >> 2;
            const int q   = rem & 3;
            int row = h + r - 1;
            row = row < 0 ? 0 : (row > HH - 1 ? HH - 1 : row);  // clamp; zeroed on use
            const _Float16* gp = xh + ((((size_t)b * HH + row) * WW + w) * CCH
                                       + ci0 + q * 8);
            copy_b128_to_lds(gp, &strip[buf][r][w][q * 8]);
        }
    };

    stage(0, 0);   // prologue

    for (int c = 0; c < 4; ++c) {
        const int ci0 = c * 32;
        wait_async_lds();    // my DMA done
        __syncthreads();     // everyone's DMA done + prev compute done
        if (c < 3) stage((c + 1) & 1, ci0 + 32);   // overlap next stage
        const _Float16 (*S)[WW][32] = strip[c & 1];

        #pragma unroll
        for (int tap = 0; tap < NTAP; ++tap) {
            const int kh = tap / 3;
            const int kw = tap % 3;

            // A fragments for two co sub-blocks (ISA f16 A layout)
            AFrag A0, A1;
            const _Float16* Ap =
                Ah + ((((size_t)b * NTAP + tap) * CCH + (cob + m)) * CCH + ci0);
            A0.u[0] = *(const v4u*)(Ap + hlf * 8);
            A0.u[1] = *(const v4u*)(Ap + 16 + hlf * 8);
            Ap += (size_t)16 * CCH;
            A1.u[0] = *(const v4u*)(Ap + hlf * 8);
            A1.u[1] = *(const v4u*)(Ap + 16 + hlf * 8);

            #pragma unroll
            for (int j = 0; j < 2; ++j) {
                const int col  = pb + j * 16 + m;
                const int wsrc = col + kw - 1;
                const bool winb  = ((unsigned)wsrc < (unsigned)WW);
                const bool valid = winb && rok[kh];
                const int  wc = winb ? wsrc : 0;
                AFrag B;
                v4u b0 = *(const v4u*)&S[kh][wc][hlf * 8];
                v4u b1 = *(const v4u*)&S[kh][wc][16 + hlf * 8];
                B.u[0] = valid ? b0 : vzero;
                B.u[1] = valid ? b1 : vzero;

                acc[0][j] = __builtin_amdgcn_wmma_f32_16x16x32_f16(
                    false, A0.h, false, B.h, (short)0, acc[0][j], false, false);
                acc[1][j] = __builtin_amdgcn_wmma_f32_16x16x32_f16(
                    false, A1.h, false, B.h, (short)0, acc[1][j], false, false);
            }
        }
    }

    // epilogue: C/D layout — VGPR r: lanes0-15 M=r, lanes16-31 M=8+r
    #pragma unroll
    for (int i = 0; i < 2; ++i) {
        #pragma unroll
        for (int j = 0; j < 2; ++j) {
            const int wout = pb + j * 16 + m;
            #pragma unroll
            for (int r = 0; r < 8; ++r) {
                const int co = cob + i * 16 + hlf * 8 + r;
                out[(((size_t)b * CCH + co) * HH + h) * WW + wout] = acc[i][j][r];
            }
        }
    }
}

// ---------------------------------------------------------------------------
extern "C" void kernel_launch(void* const* d_in, const int* in_sizes, int n_in,
                              void* d_out, int out_size, void* d_ws, size_t ws_size,
                              hipStream_t stream) {
    const float* x       = (const float*)d_in[0]; // [32,128,64,64]
    const float* kernels = (const float*)d_in[1]; // [4,128,128,3,3]
    const float* fc_w    = (const float*)d_in[2]; // [4,128]
    const float* fc_b    = (const float*)d_in[3]; // [4]
    float* out = (float*)d_out;                   // [32,128,64,64]

    float*    coeffs = (float*)d_ws;
    _Float16* Ah = (_Float16*)((char*)d_ws + 1024);
    const size_t ahElems = (size_t)BATCH * NTAP * CCH * CCH;
    _Float16* xh = Ah + ahElems;

    dyn_routing_kernel<<<BATCH, 256, 0, stream>>>(x, fc_w, fc_b, coeffs);
    dyn_mix_kernel<<<(unsigned)(ahElems / 256), 256, 0, stream>>>(kernels, coeffs, Ah);

    const size_t xElems = (size_t)BATCH * CCH * HH * WW;
    dyn_xcvt_kernel<<<(unsigned)(xElems / 256), 256, 0, stream>>>(x, xh);

    dyn_conv_wmma_kernel<<<BATCH * HH, 256, 0, stream>>>(xh, Ah, out);
    (void)in_sizes; (void)n_in; (void)out_size; (void)ws_size;
}
